// MRCGNN_sub_23407571763711
// MI455X (gfx1250) — matrix operand order repaired and checked by reading, not compiled, not run on hardware
//
#include <hip/hip_runtime.h>
#include <math.h>

// ---------------------------------------------------------------------------
// Problem constants (match reference)
// ---------------------------------------------------------------------------
#define NN 50000
#define FF 128
#define HH 2
#define DD 64
#define RR 4
#define EE 150000

typedef __attribute__((ext_vector_type(16))) __bf16 v16bf;
typedef __attribute__((ext_vector_type(8)))  float  v8f;

union FragAB { uint4 u[2]; v16bf v; };

// round-to-nearest-even f32 -> bf16 (bit trick)
__device__ __forceinline__ unsigned short f2bf(float f) {
    unsigned u = __float_as_uint(f);
    unsigned r = u + 0x7FFFu + ((u >> 16) & 1u);
    return (unsigned short)(r >> 16);
}

// order-preserving float <-> uint encoding for atomicMax-based segment max
__device__ __forceinline__ unsigned fenc(float f) {
    unsigned b = __float_as_uint(f);
    return (b & 0x80000000u) ? ~b : (b | 0x80000000u);
}
__device__ __forceinline__ float fdec(unsigned e) {
    return __uint_as_float((e & 0x80000000u) ? (e ^ 0x80000000u) : ~e);
}

// ---------------------------------------------------------------------------
// Generic WMMA GEMM:  out[m, col_ofs+n] = EPI( sum_k A'[m,k] * B'[k,n] )
//   A' : AMODE 0 -> A[m*lda + k] ;  AMODE 1 -> gelu(out_un[m,k]/denom[m, k/64])
//   B' : TRANSB ? W[n*ldb + k] : B[k*ldb + n]
//   EPI: 0 none | 1 +bias,relu | 2 +bias | 3 +bias, beta*v + (1-beta)*xprev
//   BATCH: blockIdx.y = r*H+h batches the (r,h) head GEMMs (K=N=64)
// Block: 256 thr = 8 waves; block tile = 128 rows x NCOLS cols; wave = 16 rows.
// ---------------------------------------------------------------------------
template<int KDIM, int NCOLS, int AMODE, int EPI, bool TRANSB, bool BATCH>
__global__ __launch_bounds__(256)
void wmma_gemm_k(const float* __restrict__ A, int lda,
                 const float* __restrict__ Bsrc, int ldb,
                 const float* __restrict__ bias,
                 const float* __restrict__ denom,
                 const float* __restrict__ xprev,
                 const float* __restrict__ skipv, int skip_idx,
                 float* __restrict__ out, int ldo, int nrows)
{
    constexpr int NT = NCOLS / 16;
    constexpr int AP = 40;          // padded A row (shorts): 80B, 16B-aligned
    constexpr int BP = NCOLS + 8;   // padded B row (shorts): 16B-aligned

    int col_ofs = 0;
    if constexpr (BATCH) {
        int rh = blockIdx.y;                       // r*H + h
        A    += (rh & 1) * 64;                     // head slice of k/v
        Bsrc += (size_t)rh * KDIM * NCOLS;         // a_rel[r][h]
        out  += (size_t)(rh >> 1) * (size_t)nrows * (size_t)ldo;  // k_r[r]
        col_ofs = (rh & 1) * 64;
    }

    __shared__ __attribute__((aligned(16))) unsigned short Asm[128][AP];
    __shared__ __attribute__((aligned(16))) unsigned short Bsm[KDIM][BP];

    const int tid  = threadIdx.x;
    const int lane = tid & 31;
    const int wid  = tid >> 5;
    const int wrow = wid * 16;
    const long blockRow = (long)blockIdx.x * 128;

    // ---- stage B (whole KDIM x NCOLS weight tile), fp32 -> bf16 ----
    for (int i = tid; i < KDIM * NCOLS; i += 256) {
        int k, n;
        if constexpr (TRANSB) { k = i % KDIM; n = i / KDIM; }   // coalesced W[n][k]
        else                  { n = i % NCOLS; k = i / NCOLS; } // coalesced B[k][n]
        float v = TRANSB ? Bsrc[(size_t)n * ldb + k] : Bsrc[(size_t)k * ldb + n];
        Bsm[k][n] = f2bf(v);
    }

    v8f acc[NT] = {};
    const int m_ = lane & 15;
    const int kb = (lane >> 4) * 8;

    for (int ks = 0; ks < KDIM / 32; ++ks) {
        __syncthreads();   // previous compute done before Asm restage
        // ---- stage 128x32 A chunk ----
        for (int i = tid; i < 128 * 32; i += 256) {
            int row = i >> 5;
            int kk  = i & 31;
            long grow = blockRow + row;
            float v = 0.f;
            if (grow < nrows) {
                int gk = ks * 32 + kk;
                if constexpr (AMODE == 1) {
                    float d = denom[grow * 2 + (gk >> 6)];
                    float t = (d > 0.f) ? A[grow * 128 + gk] / d : 0.f;
                    v = 0.5f * t * (1.f + erff(t * 0.70710678118654752f)); // exact gelu
                } else {
                    v = A[grow * (long)lda + gk];
                }
            }
            Asm[row][kk] = f2bf(v);
        }
        __syncthreads();   // staged data (incl. B on first iter) visible

        // A fragment: lane = row M, K halves {kb..kb+7} and {kb+16..kb+23}
        FragAB af;
        af.u[0] = *(const uint4*)&Asm[wrow + m_][kb];
        af.u[1] = *(const uint4*)&Asm[wrow + m_][kb + 16];
        const int brow = ks * 32 + kb;   // B fragment: lane = N col, K halves as A

        #pragma unroll
        for (int nt = 0; nt < NT; ++nt) {
            FragAB bf;
            #pragma unroll
            for (int s = 0; s < 16; ++s)
                ((unsigned short*)&bf)[s] = Bsm[brow + (s < 8 ? s : s + 8)][nt * 16 + m_];
            acc[nt] = __builtin_amdgcn_wmma_f32_16x16x32_bf16(
                false, af.v, false, bf.v, (short)0, acc[nt], false, false);
        }
    }

    // ---- epilogue + store (C layout: VGPR v -> row (lane>=16)*8+v, col lane&15)
    float beta = 0.f;
    if constexpr (EPI == 3) {
        float sv = skipv[skip_idx];
        beta = 1.f / (1.f + __expf(-sv));
    }
    const int rhalf = (lane >> 4) * 8;
    #pragma unroll
    for (int nt = 0; nt < NT; ++nt) {
        int gcol = nt * 16 + m_;
        #pragma unroll
        for (int vi = 0; vi < 8; ++vi) {
            long grow = blockRow + wrow + rhalf + vi;
            if (grow < nrows) {
                float v = acc[nt][vi];
                if constexpr (EPI >= 1) v += bias[gcol];
                if constexpr (EPI == 1) v = fmaxf(v, 0.f);
                if constexpr (EPI == 3) v = beta * v + (1.f - beta) * xprev[grow * 128 + gcol];
                out[grow * (long)ldo + col_ofs + gcol] = v;
            }
        }
    }
}

// ---------------------------------------------------------------------------
// Edge pass 1: alpha[r,e,h] = <q[dst,h,:], k_r[r,src,h,:]> * p_rel[r,h] / 8
//              + segment max into menc[dst,h].  One wave per edge.
// ---------------------------------------------------------------------------
__global__ __launch_bounds__(256)
void edge_alpha_k(const int* __restrict__ ei, const float* __restrict__ q,
                  const float* __restrict__ k_r, const float* __restrict__ prel,
                  float* __restrict__ alpha, unsigned* __restrict__ menc,
                  int Ecnt, int Nnodes)
{
    int g = blockIdx.x * 8 + (threadIdx.x >> 5);
    if (g >= RR * Ecnt) return;
    int lane = threadIdx.x & 31;
    int r = g / Ecnt, e = g - r * Ecnt;
    int src = ei[(r * 2 + 0) * Ecnt + e];
    int dst = ei[(r * 2 + 1) * Ecnt + e];

    float4 qv = ((const float4*)(q + (size_t)dst * 128))[lane];
    float4 kv = ((const float4*)(k_r + ((size_t)r * Nnodes + src) * 128))[lane];
    float s = qv.x * kv.x + qv.y * kv.y + qv.z * kv.z + qv.w * kv.w;
    s += __shfl_xor(s, 1); s += __shfl_xor(s, 2);
    s += __shfl_xor(s, 4); s += __shfl_xor(s, 8);     // sum over 16-lane head group

    if ((lane & 15) == 0) {
        int h = lane >> 4;
        float al = s * prel[r * 2 + h] * 0.125f;       // / sqrt(64)
        alpha[(size_t)g * 2 + h] = al;
        atomicMax(&menc[(size_t)dst * 2 + h], fenc(al));
    }
}

// ---------------------------------------------------------------------------
// Edge pass 2: a = exp(alpha - m[dst]); denom[dst,h]+=a;
//              out_un[dst,:] += a * v_r[r,src,:]   (unnormalized message)
// ---------------------------------------------------------------------------
__global__ __launch_bounds__(256)
void edge_accum_k(const int* __restrict__ ei, const float* __restrict__ v_r,
                  const float* __restrict__ alpha, const unsigned* __restrict__ menc,
                  float* __restrict__ denom, float* __restrict__ out_un,
                  int Ecnt, int Nnodes)
{
    int g = blockIdx.x * 8 + (threadIdx.x >> 5);
    if (g >= RR * Ecnt) return;
    int lane = threadIdx.x & 31;
    int r = g / Ecnt, e = g - r * Ecnt;
    int src = ei[(r * 2 + 0) * Ecnt + e];
    int dst = ei[(r * 2 + 1) * Ecnt + e];
    int h = lane >> 4;

    float al = alpha[(size_t)g * 2 + h];
    float m  = fdec(menc[(size_t)dst * 2 + h]);
    float a  = expf(al - m);
    if ((lane & 15) == 0) atomicAdd(&denom[(size_t)dst * 2 + h], a);

    float4 vv = ((const float4*)(v_r + ((size_t)r * Nnodes + src) * 128))[lane];
    float* op = out_un + (size_t)dst * 128 + lane * 4;
    atomicAdd(op + 0, vv.x * a);
    atomicAdd(op + 1, vv.y * a);
    atomicAdd(op + 2, vv.z * a);
    atomicAdd(op + 3, vv.w * a);
}

__global__ void zero_k(unsigned* __restrict__ p, long n) {
    long i = (long)blockIdx.x * blockDim.x + threadIdx.x;
    long stride = (long)gridDim.x * blockDim.x;
    for (; i < n; i += stride) p[i] = 0u;
}

__global__ __launch_bounds__(256)
void mean_k(const float* __restrict__ x2, float* __restrict__ h, int n) {
    __shared__ float red[256];
    int f = blockIdx.x;
    float s = 0.f;
    for (int i = threadIdx.x; i < n; i += 256) s += x2[(size_t)i * 128 + f];
    red[threadIdx.x] = s;
    __syncthreads();
    for (int off = 128; off > 0; off >>= 1) {
        if (threadIdx.x < off) red[threadIdx.x] += red[threadIdx.x + off];
        __syncthreads();
    }
    if (threadIdx.x == 0) h[f] = red[0] / (float)n;
}

// ---------------------------------------------------------------------------
extern "C" void kernel_launch(void* const* d_in, const int* in_sizes, int n_in,
                              void* d_out, int out_size, void* d_ws, size_t ws_size,
                              hipStream_t stream) {
    const float* x     = (const float*)d_in[0];
    const int*   ei    = (const int*)  d_in[1];
    const float* W_lin = (const float*)d_in[2];
    const float* b_lin = (const float*)d_in[3];
    const float* Wk    = (const float*)d_in[4];
    const float* bk    = (const float*)d_in[5];
    const float* Wq    = (const float*)d_in[6];
    const float* bq    = (const float*)d_in[7];
    const float* Wv    = (const float*)d_in[8];
    const float* bv    = (const float*)d_in[9];
    const float* Wa    = (const float*)d_in[10];
    const float* ba    = (const float*)d_in[11];
    const float* a_rel = (const float*)d_in[12];
    const float* m_rel = (const float*)d_in[13];
    const float* p_rel = (const float*)d_in[14];
    const float* skip  = (const float*)d_in[15];

    const long NF = (long)NN * FF;               // 6.4M floats
    float*    ws    = (float*)d_ws;
    float*    x0    = ws;
    float*    kbuf  = x0   + NF;
    float*    qbuf  = kbuf + NF;
    float*    vbuf  = qbuf + NF;
    float*    k_r   = vbuf + NF;                 // R*N*128
    float*    v_r   = k_r  + (long)RR * NF;
    float*    alpha = v_r  + (long)RR * NF;      // R*E*H
    unsigned* menc  = (unsigned*)(alpha + (long)RR * EE * HH);
    float*    denom = (float*)(menc + (long)NN * HH);
    float*    outun = denom + (long)NN * HH;

    float* outf = (float*)d_out;
    float* hvec = outf;
    float* x1   = outf + FF;
    float* x2   = x1 + NF;

    const int NBLK = (NN + 127) / 128;           // 391
    const int EBLK = (RR * EE) / 8;              // 75000

    // Stage 0: x0 = relu(x @ W_lin^T + b_lin)
    wmma_gemm_k<128,128,0,1,true,false><<<NBLK, 256, 0, stream>>>(
        x, 128, W_lin, 128, b_lin, nullptr, nullptr, nullptr, 0, x0, 128, NN);

    const float* lin = x0;
    float* lout[2] = { x1, x2 };
    for (int l = 0; l < 2; ++l) {
        const float* Wk_l = Wk + (long)l * FF * FF;  const float* bk_l = bk + l * FF;
        const float* Wq_l = Wq + (long)l * FF * FF;  const float* bq_l = bq + l * FF;
        const float* Wv_l = Wv + (long)l * FF * FF;  const float* bv_l = bv + l * FF;
        const float* Wa_l = Wa + (long)l * FF * FF;  const float* ba_l = ba + l * FF;
        const float* ar_l = a_rel + (long)l * RR * HH * DD * DD;
        const float* mr_l = m_rel + (long)l * RR * HH * DD * DD;
        const float* pr_l = p_rel + l * RR * HH;

        // k/q/v projections (bias, no act)
        wmma_gemm_k<128,128,0,2,true,false><<<NBLK, 256, 0, stream>>>(
            lin, 128, Wk_l, 128, bk_l, nullptr, nullptr, nullptr, 0, kbuf, 128, NN);
        wmma_gemm_k<128,128,0,2,true,false><<<NBLK, 256, 0, stream>>>(
            lin, 128, Wq_l, 128, bq_l, nullptr, nullptr, nullptr, 0, qbuf, 128, NN);
        wmma_gemm_k<128,128,0,2,true,false><<<NBLK, 256, 0, stream>>>(
            lin, 128, Wv_l, 128, bv_l, nullptr, nullptr, nullptr, 0, vbuf, 128, NN);

        // relation-transformed keys/values: k_r[r,n,h,:] = k[n,h,:] @ a_rel[r,h]
        wmma_gemm_k<64,64,0,0,false,true><<<dim3(NBLK, RR * HH), 256, 0, stream>>>(
            kbuf, 128, ar_l, 64, nullptr, nullptr, nullptr, nullptr, 0, k_r, 128, NN);
        wmma_gemm_k<64,64,0,0,false,true><<<dim3(NBLK, RR * HH), 256, 0, stream>>>(
            vbuf, 128, mr_l, 64, nullptr, nullptr, nullptr, nullptr, 0, v_r, 128, NN);

        // zero menc + denom + out_un (contiguous: N*2 + N*2 + N*128 words)
        zero_k<<<1024, 256, 0, stream>>>(menc, (long)NN * (HH + HH + FF));

        // segment softmax + message aggregation
        edge_alpha_k<<<EBLK, 256, 0, stream>>>(ei, qbuf, k_r, pr_l, alpha, menc, EE, NN);
        edge_accum_k<<<EBLK, 256, 0, stream>>>(ei, v_r, alpha, menc, denom, outun, EE, NN);

        // out = beta * (gelu(out_un/denom) @ Wa^T + ba) + (1-beta) * lin
        wmma_gemm_k<128,128,1,3,true,false><<<NBLK, 256, 0, stream>>>(
            outun, 128, Wa_l, 128, ba_l, denom, lin, skip, l, lout[l], 128, NN);

        lin = lout[l];
    }

    // h = mean(x2, axis=0)
    mean_k<<<FF, 256, 0, stream>>>(x2, hvec, NN);
}
